// SpatioTemporalGCN_58858231824835
// MI455X (gfx1250) — matrix operand order repaired and checked by reading, not compiled, not run on hardware
//
#include <hip/hip_runtime.h>

// SpatioTemporalGCN (2x NNConv + readout) for gfx1250.
// Edge-conditioned weight GEMMs refactored as bilinear GEMMs
//   msg = (h_mlp(e) outer x_src(e)) @ reshaped(w2)
// so the 1 GB per-edge weight tensor is never materialized.  Inner loops use
// v_wmma_f32_16x16x32_bf16 with f32 accumulation; A fragments are produced
// with packed v_pk_mul_bf16 (h-scalar broadcast x preloaded x-pattern).

#define N_NODES 10000
#define N_EDGES 65536
#define C_IN    16
#define C_H     64

// conv1: K_u = 16*64 = 1024, + 16 bias rows -> pad to 33 chunks of 32
// conv2: K_u = 64*64 = 4096, + 64 bias rows -> exactly 130 chunks of 32
#define KU1   1024
#define NCH1  33
#define KU2   4096
#define NCH2  130

typedef __attribute__((ext_vector_type(16))) __bf16          v16bf;
typedef __attribute__((ext_vector_type(8)))  unsigned int    v8u;
typedef __attribute__((ext_vector_type(8)))  float           v8f;

static __device__ __forceinline__ unsigned short f2bf(float f) {
  unsigned int b = __builtin_bit_cast(unsigned int, f);
  b += 0x7FFFu + ((b >> 16) & 1u);        // round-to-nearest-even
  return (unsigned short)(b >> 16);
}

// packed bf16 multiply (VOP3P V_PK_MUL_BF16): two bf16 lanes per dword
static __device__ __forceinline__ unsigned int pkmul_bf16(unsigned int a,
                                                          unsigned int b) {
  unsigned int d;
  asm("v_pk_mul_bf16 %0, %1, %2" : "=v"(d) : "v"(a), "v"(b));
  return d;
}

static __device__ __forceinline__ v8f wmma_bf16(v8u a, v8u b, v8f c) {
  return __builtin_amdgcn_wmma_f32_16x16x32_bf16(
      false, __builtin_bit_cast(v16bf, a),
      false, __builtin_bit_cast(v16bf, b),
      (short)0, c, false, false);
}

// ---------------------------------------------------------------- utilities
__global__ void k_zero(float* __restrict__ p, int n) {
  int i = blockIdx.x * 256 + threadIdx.x;
  if (i < n) p[i] = 0.0f;
}

// hidden = relu(edge_attr @ w1 + b1)  -> bf16 [E, 64]
__global__ void k_edge_mlp(const float* __restrict__ ea,
                           const float* __restrict__ w1,
                           const float* __restrict__ b1,
                           unsigned short* __restrict__ out) {
  int idx = blockIdx.x * 256 + threadIdx.x;
  if (idx >= N_EDGES * C_H) return;
  int e = idx >> 6, m = idx & 63;
  const float* a = ea + (size_t)e * 4;
  float s = b1[m];
  s = fmaf(a[0], w1[0 * C_H + m], s);
  s = fmaf(a[1], w1[1 * C_H + m], s);
  s = fmaf(a[2], w1[2 * C_H + m], s);
  s = fmaf(a[3], w1[3 * C_H + m], s);
  out[idx] = f2bf(fmaxf(s, 0.0f));
}

// gather source-node features to edge-major bf16 [E, CIN]
template <int CIN>
__global__ void k_gather(const float* __restrict__ feat,
                         const int* __restrict__ srci,
                         unsigned short* __restrict__ out) {
  int idx = blockIdx.x * 256 + threadIdx.x;
  if (idx >= N_EDGES * CIN) return;
  int e = idx / CIN, i = idx & (CIN - 1);
  out[idx] = f2bf(feat[(size_t)srci[e] * CIN + i]);
}

// Pre-swizzle B = [w2 reshaped (m*CIN+i, o) ; b2 rows ; zero pad] into the
// per-(chunk, n-tile, lane) bf16 fragment layout consumed by the WMMA loop.
// 16-bit B (32x16) K map: K = (j&7) + (j>>3)*16 + (lane>>4)*8.
template <int CIN, int KU, int NCHUNK>
__global__ void k_bprep(const float* __restrict__ w2,
                        const float* __restrict__ b2,
                        unsigned short* __restrict__ bfrag) {
  int idx = blockIdx.x * 256 + threadIdx.x;
  if (idx >= NCHUNK * 4 * 32 * 16) return;
  int j    = idx & 15;
  int lane = (idx >> 4) & 31;
  int t    = (idx >> 9) & 3;
  int c    = idx >> 11;
  int n = t * 16 + (lane & 15);
  int k = c * 32 + (j & 7) + ((j >> 3) << 4) + ((lane >> 4) << 3);
  float v = 0.0f;
  if (k < KU) {
    int m = k / CIN, i = k % CIN;
    v = w2[(size_t)m * (CIN * C_H) + i * C_H + n];
  } else if (k < KU + CIN) {
    v = b2[(k - KU) * C_H + n];
  }
  bfrag[idx] = f2bf(v);
}

// in-degree counts for the scatter-mean
__global__ void k_count(const int* __restrict__ dst, float* __restrict__ cnt) {
  int e = blockIdx.x * 256 + threadIdx.x;
  if (e < N_EDGES) atomicAdd(cnt + dst[e], 1.0f);
}

// ------------------------------------------------ main WMMA message kernel
// One workgroup = 16 edges x 64 outputs; 4 wave32s, each owning a 16-wide N
// slice.  A (16x32 bf16) = outer product h_e[m]*x_e[i]: since m is constant
// per K-chunk and the x access pattern is chunk-invariant, A-build reduces to
// 8 v_pk_mul_bf16 against preloaded packed X registers.  Bias rows of the
// extended GEMM are the X registers themselves.  B fragments are pre-swizzled
// in global memory (L2-resident) and double-buffered.
template <int CIN, int KU, int NCHUNK>
__global__ void __launch_bounds__(128)
k_nnconv_msg(const unsigned short* __restrict__ hedge,  // [E,64] bf16
             const unsigned short* __restrict__ xg,     // [E,CIN] bf16
             const unsigned short* __restrict__ bfrag,  // fragments bf16
             const int* __restrict__ dst,               // [E]
             float* __restrict__ agg) {                 // [N,64] f32 (atomic)
  constexpr int HS = 66;                    // padded LDS row stride (shorts)
  __shared__ unsigned short h_lds[16 * HS];
  __shared__ unsigned short x_lds[16 * HS];
  __shared__ long doff[16];                 // dst byte offsets

  const int e0   = blockIdx.x << 4;
  const int tid  = threadIdx.x;
  const int wave = tid >> 5;                // n-tile 0..3
  const int lane = tid & 31;

  for (int idx = tid; idx < 16 * C_H; idx += 128) {
    int e = idx >> 6, m = idx & 63;
    h_lds[e * HS + m] = hedge[(size_t)(e0 + e) * C_H + m];
  }
  for (int idx = tid; idx < 16 * CIN; idx += 128) {
    int e = idx / CIN, i = idx & (CIN - 1);
    x_lds[e * HS + i] = xg[(size_t)(e0 + e) * CIN + i];
  }
  if (tid < 16) doff[tid] = (long)dst[e0 + tid] * (C_H * 4);
  __syncthreads();

  const int row = lane & 15;                // A-matrix row (edge within tile)
  const int hi  = lane >> 4;                // lane-group half
  const unsigned short* hrow = h_lds + row * HS;
  const unsigned short* xrow = x_lds + row * HS;

  // Preload per-lane x pattern.  16-bit A (16x32) K map:
  //   low(j) = (j&7) + (j>>3)*16 + hi*8
  // conv2: i = (c&1)*32 + low(j)   -> X0 (even chunks), X1 (odd chunks)
  // conv1: i = (j&7) + hi*8        -> X0 (all chunks); X1 = bias-tail frag
  v8u X0, X1;
#pragma unroll
  for (int p = 0; p < 8; ++p) {
    const int j0 = 2 * p, j1 = 2 * p + 1;
    if constexpr (CIN == 64) {
      const int l0 = (j0 & 7) + ((j0 >> 3) << 4) + (hi << 3);
      const int l1 = (j1 & 7) + ((j1 >> 3) << 4) + (hi << 3);
      X0[p] = (unsigned)xrow[l0] | ((unsigned)xrow[l1] << 16);
      X1[p] = (unsigned)xrow[32 + l0] | ((unsigned)xrow[32 + l1] << 16);
    } else {
      const int i0 = (j0 & 7) + (hi << 3);
      const int i1 = (j1 & 7) + (hi << 3);
      X0[p] = (unsigned)xrow[i0] | ((unsigned)xrow[i1] << 16);
      X1[p] = (p < 4) ? X0[p] : 0u;         // bias rows then zero pad
    }
  }

  const unsigned short* bbase =
      bfrag + (((size_t)wave * 32 + lane) << 4);
  auto loadB = [&](int c) -> v8u {
    return *(const v8u*)(bbase + ((size_t)c << 11));   // c*4*32*16 shorts
  };

  v8f acc = {};
  if constexpr (CIN == 64) {
    // 64 chunk-pairs sharing one h scalar, then 2 bias-tail chunks.
    v8u b0 = loadB(0), b1 = loadB(1);
    for (int cp = 0; cp < KU / 64; ++cp) {
      v8u nb0 = loadB(2 * cp + 2);          // cp==63 prefetches the tail
      v8u nb1 = loadB(2 * cp + 3);
      const unsigned int hp = (unsigned int)hrow[cp] * 0x10001u;
      v8u a0, a1;
#pragma unroll
      for (int p = 0; p < 8; ++p) {
        a0[p] = pkmul_bf16(hp, X0[p]);
        a1[p] = pkmul_bf16(hp, X1[p]);
      }
      acc = wmma_bf16(a0, b0, acc);
      acc = wmma_bf16(a1, b1, acc);
      b0 = nb0; b1 = nb1;
    }
    acc = wmma_bf16(X0, b0, acc);           // bias rows k-KU in [0,32)
    acc = wmma_bf16(X1, b1, acc);           // bias rows k-KU in [32,64)
  } else {
    // 32 main chunks (two h scalars each), then 1 bias+pad tail chunk.
    v8u b0 = loadB(0);
    for (int c = 0; c < KU / 32; ++c) {
      v8u nb = loadB(c + 1);                // c==31 prefetches the tail
      const unsigned int h0 = (unsigned int)hrow[2 * c] * 0x10001u;
      const unsigned int h1 = (unsigned int)hrow[2 * c + 1] * 0x10001u;
      v8u a;
#pragma unroll
      for (int p = 0; p < 4; ++p) a[p] = pkmul_bf16(h0, X0[p]);
#pragma unroll
      for (int p = 4; p < 8; ++p) a[p] = pkmul_bf16(h1, X0[p]);
      acc = wmma_bf16(a, b0, acc);
      b0 = nb;
    }
    acc = wmma_bf16(X1, b0, acc);           // bias rows + zero pad
  }

  // C/D layout: VGPR r, lanes 0-15 -> M=r, lanes 16-31 -> M=r+8; N = lane%16
  const int n = (wave << 4) + row;
#pragma unroll
  for (int r = 0; r < 8; ++r) {
    const int m = r + (hi << 3);            // edge within tile
    atomicAdd((float*)((char*)agg + doff[m] + n * 4), acc[r]);
  }
}

// out = relu(agg/max(cnt,1) + x @ root + bias)
template <int CIN>
__global__ void k_combine(const float* __restrict__ agg,
                          const float* __restrict__ cnt,
                          const float* __restrict__ xin,
                          const float* __restrict__ root,
                          const float* __restrict__ bias,
                          float* __restrict__ out) {
  int idx = blockIdx.x * 256 + threadIdx.x;
  if (idx >= N_NODES * C_H) return;
  int n = idx >> 6, o = idx & 63;
  float s = agg[idx] / fmaxf(cnt[n], 1.0f) + bias[o];
  const float* xr = xin + (size_t)n * CIN;
#pragma unroll 8
  for (int i = 0; i < CIN; ++i) s = fmaf(xr[i], root[i * C_H + o], s);
  out[idx] = fmaxf(s, 0.0f);
}

// readout: relu(h @ lin1 + b1) @ lin2 + b2 -> [N]
__global__ void k_readout(const float* __restrict__ h2,
                          const float* __restrict__ l1w,
                          const float* __restrict__ l1b,
                          const float* __restrict__ l2w,
                          const float* __restrict__ l2b,
                          float* __restrict__ out) {
  int n = blockIdx.x * 256 + threadIdx.x;
  if (n >= N_NODES) return;
  float t[8];
#pragma unroll
  for (int j = 0; j < 8; ++j) t[j] = l1b[j];
  const float* hr = h2 + (size_t)n * C_H;
  for (int i = 0; i < C_H; ++i) {
    float hv = hr[i];
#pragma unroll
    for (int j = 0; j < 8; ++j) t[j] = fmaf(hv, l1w[i * 8 + j], t[j]);
  }
  float s = l2b[0];
#pragma unroll
  for (int j = 0; j < 8; ++j) s = fmaf(fmaxf(t[j], 0.0f), l2w[j], s);
  out[n] = s;
}

// ------------------------------------------------------------------- launch
extern "C" void kernel_launch(void* const* d_in, const int* in_sizes, int n_in,
                              void* d_out, int out_size, void* d_ws,
                              size_t ws_size, hipStream_t stream) {
  (void)in_sizes; (void)n_in; (void)out_size; (void)ws_size;
  const float* x     = (const float*)d_in[0];
  const int*   ei    = (const int*)d_in[1];   // [2,E] int32
  const float* ea    = (const float*)d_in[2];
  const float* n1w1  = (const float*)d_in[3];
  const float* n1b1  = (const float*)d_in[4];
  const float* n1w2  = (const float*)d_in[5];
  const float* n1b2  = (const float*)d_in[6];
  const float* root1 = (const float*)d_in[7];
  const float* bias1 = (const float*)d_in[8];
  const float* n2w1  = (const float*)d_in[9];
  const float* n2b1  = (const float*)d_in[10];
  const float* n2w2  = (const float*)d_in[11];
  const float* n2b2  = (const float*)d_in[12];
  const float* root2 = (const float*)d_in[13];
  const float* bias2 = (const float*)d_in[14];
  const float* l1w   = (const float*)d_in[15];
  const float* l1b   = (const float*)d_in[16];
  const float* l2w   = (const float*)d_in[17];
  const float* l2b   = (const float*)d_in[18];
  const int* srci = ei;
  const int* dsti = ei + N_EDGES;

  char* ws = (char*)d_ws;
  size_t off = 0;
  auto alloc = [&](size_t bytes) -> char* {
    char* p = ws + off;
    off = (off + bytes + 255) & ~(size_t)255;
    return p;
  };
  // agg1/agg2/cnt contiguous (sizes 256B-aligned) -> single zero pass
  float* agg1 = (float*)alloc((size_t)N_NODES * C_H * 4);
  float* agg2 = (float*)alloc((size_t)N_NODES * C_H * 4);
  float* cnt  = (float*)alloc((size_t)N_NODES * 4);
  float* h1   = (float*)alloc((size_t)N_NODES * C_H * 4);
  float* h2   = (float*)alloc((size_t)N_NODES * C_H * 4);
  unsigned short* he1 = (unsigned short*)alloc((size_t)N_EDGES * C_H * 2);
  unsigned short* he2 = (unsigned short*)alloc((size_t)N_EDGES * C_H * 2);
  unsigned short* xg1 = (unsigned short*)alloc((size_t)N_EDGES * C_IN * 2);
  unsigned short* xg2 = (unsigned short*)alloc((size_t)N_EDGES * C_H * 2);
  unsigned short* bf1 = (unsigned short*)alloc((size_t)NCH1 * 4 * 32 * 16 * 2);
  unsigned short* bf2 = (unsigned short*)alloc((size_t)NCH2 * 4 * 32 * 16 * 2);

  const int B = 256;
  // 1. zero accumulators + counts
  {
    int n = 2 * N_NODES * C_H + N_NODES;
    k_zero<<<(n + B - 1) / B, B, 0, stream>>>(agg1, n);
  }
  // 2. pre-swizzle B matrices (bf16 fragments)
  k_bprep<C_IN, KU1, NCH1><<<(NCH1 * 2048 + B - 1) / B, B, 0, stream>>>(n1w2, n1b2, bf1);
  k_bprep<C_H, KU2, NCH2><<<(NCH2 * 2048 + B - 1) / B, B, 0, stream>>>(n2w2, n2b2, bf2);
  // 3. edge MLP hidden layers
  k_edge_mlp<<<(N_EDGES * C_H + B - 1) / B, B, 0, stream>>>(ea, n1w1, n1b1, he1);
  k_edge_mlp<<<(N_EDGES * C_H + B - 1) / B, B, 0, stream>>>(ea, n2w1, n2b1, he2);
  // 4. gather x[src] for conv1; in-degree counts
  k_gather<C_IN><<<(N_EDGES * C_IN + B - 1) / B, B, 0, stream>>>(x, srci, xg1);
  k_count<<<(N_EDGES + B - 1) / B, B, 0, stream>>>(dsti, cnt);
  // 5. conv1 messages (WMMA) + scatter-add
  k_nnconv_msg<C_IN, KU1, NCH1><<<N_EDGES / 16, 128, 0, stream>>>(he1, xg1, bf1, dsti, agg1);
  // 6. mean + root transform + bias + relu -> h1
  k_combine<C_IN><<<(N_NODES * C_H + B - 1) / B, B, 0, stream>>>(agg1, cnt, x, root1, bias1, h1);
  // 7. gather h1[src] for conv2
  k_gather<C_H><<<(N_EDGES * C_H + B - 1) / B, B, 0, stream>>>(h1, srci, xg2);
  // 8. conv2 messages (WMMA) + scatter-add
  k_nnconv_msg<C_H, KU2, NCH2><<<N_EDGES / 16, 128, 0, stream>>>(he2, xg2, bf2, dsti, agg2);
  // 9. mean + root transform + bias + relu -> h2
  k_combine<C_H><<<(N_NODES * C_H + B - 1) / B, B, 0, stream>>>(agg2, cnt, h1, root2, bias2, h2);
  // 10. readout MLP -> out[N]
  k_readout<<<(N_NODES + B - 1) / B, B, 0, stream>>>(h2, l1w, l1b, l2w, l2b, (float*)d_out);
}
